// NetVLAD_79018808312293
// MI455X (gfx1250) — matrix operand order repaired and checked by reading, not compiled, not run on hardware
//
#include <hip/hip_runtime.h>
#include <hip/hip_bf16.h>

typedef __attribute__((ext_vector_type(16))) _Float16 v16h;
typedef __attribute__((ext_vector_type(8)))  _Float16 v8h;
typedef __attribute__((ext_vector_type(8)))  float    v8f;

#define N_ 64
#define C_ 4096
#define D_ 128
#define K_ 64

// ---------------------------------------------------------------------------
// Kernel 1: per (image n, C-split sp) partial NetVLAD accumulation.
// grid = (S, N), block = 256 (8 wave32). Each block consumes C/S descriptors
// and emits a partial (64x128) f32 first-term accumulator + asum[64].
// ---------------------------------------------------------------------------
__global__ __launch_bounds__(256)
void netvlad_partial(const float* __restrict__ x,
                     const float* __restrict__ weight,
                     const float* __restrict__ bias,
                     const float* __restrict__ masks,
                     float* __restrict__ part,    // [N*S][64*128]
                     float* __restrict__ asump,   // [N*S][64]
                     int S)
{
    // Shared arena:
    //   [0,16384)      wfrag : f16 W in B-fragment layout [4 kd][4 nt][32 lane][16]
    //   [16384,24576)  xn    : f16 [32][128] row-major (A operand, GEMM1)
    //   [24576,32768)  xnT   : f16 [128][32] transposed (B operand, GEMM2)
    //   [32768,36864)  a_s   : f16 [64][32]  softassign^T (A operand, GEMM2)
    //   [36864,45056)  logits: f32 [32][64]
    __shared__ __align__(64) unsigned char smem_raw[45056];
    __shared__ float bias_s[64];
    __shared__ float asum_s[64];

    _Float16* wfrag  = (_Float16*)(smem_raw);
    _Float16* xn     = (_Float16*)(smem_raw + 16384);
    _Float16* xnT    = (_Float16*)(smem_raw + 24576);
    _Float16* a_s    = (_Float16*)(smem_raw + 32768);
    float*    logits = (float*)   (smem_raw + 36864);

    const int tid  = threadIdx.x;
    const int lane = tid & 31;
    const int w    = tid >> 5;
    const int sp   = blockIdx.x;
    const int n    = blockIdx.y;

    // ---- pre-pack W into per-lane-contiguous B fragments; load bias; zero asum
    for (int idx = tid; idx < 16 * 32; idx += 256) {
        int f  = idx >> 5;
        int L  = idx & 31;
        int kd = f >> 2;
        int nt = f & 3;
        int k     = nt * 16 + (L & 15);
        int dbase = kd * 32 + ((L >> 4) << 4);
        const float* wp = weight + k * D_ + dbase;
        _Float16* dst = wfrag + idx * 16;
#pragma unroll
        for (int i = 0; i < 16; ++i) dst[i] = (_Float16)wp[i];
    }
    if (tid < 64) { bias_s[tid] = bias[tid]; asum_s[tid] = 0.0f; }
    __syncthreads();

    v8f zero = {};
    v8f vacc[4] = {zero, zero, zero, zero};

    const float* xg = x + (size_t)n * (C_ * D_);
    const int CS   = C_ / S;
    const int cbeg = sp * CS;
    const int cend = cbeg + CS;

    for (int c0 = cbeg; c0 < cend; c0 += 32) {
        // ---- stage 1: load + L2-normalize 32 descriptors; f16 to LDS (2 layouts)
        {
            int row = tid >> 3;               // 0..31
            int d0  = (tid & 7) << 4;         // 16 floats per thread
            const float* xr = xg + (c0 + row) * D_ + d0;
            float4 p0 = ((const float4*)xr)[0];
            float4 p1 = ((const float4*)xr)[1];
            float4 p2 = ((const float4*)xr)[2];
            float4 p3 = ((const float4*)xr)[3];
            if (c0 + 32 < cend) __builtin_prefetch(xr + 32 * D_, 0, 1);
            float xv[16] = {p0.x,p0.y,p0.z,p0.w, p1.x,p1.y,p1.z,p1.w,
                            p2.x,p2.y,p2.z,p2.w, p3.x,p3.y,p3.z,p3.w};
            float s = 0.f;
#pragma unroll
            for (int i = 0; i < 16; ++i) s += xv[i] * xv[i];
            s += __shfl_xor(s, 1); s += __shfl_xor(s, 2); s += __shfl_xor(s, 4);
            float inv = 1.0f / fmaxf(sqrtf(s), 1e-12f);
            _Float16 h[16];
#pragma unroll
            for (int i = 0; i < 16; ++i) h[i] = (_Float16)(xv[i] * inv);
            v8h h0, h1;
#pragma unroll
            for (int i = 0; i < 8; ++i) { h0[i] = h[i]; h1[i] = h[i + 8]; }
            *(v8h*)&xn[row * 128 + d0]     = h0;
            *(v8h*)&xn[row * 128 + d0 + 8] = h1;
#pragma unroll
            for (int i = 0; i < 16; ++i) xnT[(d0 + i) * 32 + row] = h[i];
        }
        __syncthreads();

        // ---- stage 2: logits(32x64) = xn @ W^T + bias  (WMMA, 1 tile/wave)
        {
            int mt = w >> 2, nt = w & 3;
            v8f acc = {};
            int arow  = mt * 16 + (lane & 15);
            int aoff0 = (lane >> 4) << 4;
#pragma unroll
            for (int kd = 0; kd < 4; ++kd) {
                int aoff = kd * 32 + aoff0;
                v8h lo = *(const v8h*)&xn[arow * 128 + aoff];
                v8h hi = *(const v8h*)&xn[arow * 128 + aoff + 8];
                v16h A = __builtin_shufflevector(lo, hi, 0,1,2,3,4,5,6,7,8,9,10,11,12,13,14,15);
                v16h B = *(const v16h*)&wfrag[((kd * 4 + nt) * 32 + lane) * 16];
                acc = __builtin_amdgcn_wmma_f32_16x16x32_f16(
                          false, A, false, B, (short)0, acc, false, false);
            }
            int col   = nt * 16 + (lane & 15);
            float bv  = bias_s[col];
            int rbase = mt * 16 + ((lane >> 4) << 3);
#pragma unroll
            for (int r = 0; r < 8; ++r) logits[(rbase + r) * 64 + col] = acc[r] + bv;
        }
        __syncthreads();

        // ---- stage 3: masked softmax over K=64; store softassign^T (f16)
        {
            int row = tid >> 3;
            int k0  = (tid & 7) << 3;
            float lv[8];
            float m = -3.4e38f;
#pragma unroll
            for (int i = 0; i < 8; ++i) { lv[i] = logits[row * 64 + k0 + i]; m = fmaxf(m, lv[i]); }
            m = fmaxf(m, __shfl_xor(m, 1));
            m = fmaxf(m, __shfl_xor(m, 2));
            m = fmaxf(m, __shfl_xor(m, 4));
            float s = 0.f;
#pragma unroll
            for (int i = 0; i < 8; ++i) { lv[i] = __expf(lv[i] - m); s += lv[i]; }
            s += __shfl_xor(s, 1); s += __shfl_xor(s, 2); s += __shfl_xor(s, 4);
            float mk = masks[(size_t)n * C_ + c0 + row];
            float sc = mk / s;
#pragma unroll
            for (int i = 0; i < 8; ++i) a_s[(k0 + i) * 32 + row] = (_Float16)(lv[i] * sc);
        }
        __syncthreads();

        // running per-cluster assignment sums
        if (tid < 64) {
            float s = 0.f;
#pragma unroll
            for (int c = 0; c < 32; ++c) s += (float)a_s[tid * 32 + c];
            asum_s[tid] += s;
        }

        // ---- stage 4: first_term += softassign^T @ xn  (WMMA, 4 tiles/wave)
        {
            int kt = w >> 1, dh = w & 1;
            int krow = kt * 16 + (lane & 15);
            int coff = (lane >> 4) << 4;
            v8h lo = *(const v8h*)&a_s[krow * 32 + coff];
            v8h hi = *(const v8h*)&a_s[krow * 32 + coff + 8];
            v16h A = __builtin_shufflevector(lo, hi, 0,1,2,3,4,5,6,7,8,9,10,11,12,13,14,15);
#pragma unroll
            for (int j = 0; j < 4; ++j) {
                int dcol = (dh * 4 + j) * 16 + (lane & 15);
                v16h B = *(const v16h*)&xnT[dcol * 32 + ((lane >> 4) << 4)];
                vacc[j] = __builtin_amdgcn_wmma_f32_16x16x32_f16(
                              false, A, false, B, (short)0, vacc[j], false, false);
            }
        }
        __syncthreads();
    }

    // ---- stream partial accumulators to workspace (coalesced 64B segments)
    {
        float* pv = part + ((size_t)n * S + sp) * (K_ * D_);
        int kt = w >> 1, dh = w & 1;
        int rbase = kt * 16 + ((lane >> 4) << 3);
#pragma unroll
        for (int j = 0; j < 4; ++j) {
            int dcol = (dh * 4 + j) * 16 + (lane & 15);
#pragma unroll
            for (int r = 0; r < 8; ++r) pv[(rbase + r) * 128 + dcol] = vacc[j][r];
        }
        if (tid < 64) asump[((size_t)n * S + sp) * 64 + tid] = asum_s[tid];
    }
}

// ---------------------------------------------------------------------------
// Kernel 2: deterministic reduction over S partials + second term +
// intra/global L2 normalization. grid = N, block = 256.
// ---------------------------------------------------------------------------
__global__ __launch_bounds__(256)
void netvlad_reduce(const float* __restrict__ centroids,
                    const float* __restrict__ part,
                    const float* __restrict__ asump,
                    float* __restrict__ out,
                    int S)
{
    __shared__ float asum_s[64];
    __shared__ float red_s[256];

    const int tid = threadIdx.x;
    const int n   = blockIdx.x;

    if (tid < 64) {
        float s = 0.f;
        for (int q = 0; q < S; ++q) s += asump[((size_t)n * S + q) * 64 + tid];
        asum_s[tid] = s;
    }

    const int rid = tid >> 2;           // cluster row 0..63 (4 threads each)
    const int d0  = (tid & 3) << 5;     // 32 elements per thread

    float buf[32];
#pragma unroll
    for (int i = 0; i < 32; ++i) buf[i] = 0.f;
    for (int q = 0; q < S; ++q) {
        const float4* pp = (const float4*)(part + ((size_t)n * S + q) * (K_ * D_)
                                                + rid * 128 + d0);
#pragma unroll
        for (int i = 0; i < 8; ++i) {
            float4 v = pp[i];
            buf[i*4+0] += v.x; buf[i*4+1] += v.y; buf[i*4+2] += v.z; buf[i*4+3] += v.w;
        }
    }
    __syncthreads();

    float av = asum_s[rid];
    const float* cp = centroids + rid * 128 + d0;
    float s = 0.f;
#pragma unroll
    for (int i = 0; i < 32; ++i) {
        float v = buf[i] - av * cp[i];
        buf[i] = v; s += v * v;
    }
    s += __shfl_xor(s, 1); s += __shfl_xor(s, 2);
    float inv = 1.0f / fmaxf(sqrtf(s), 1e-12f);
    float gs = 0.f;
#pragma unroll
    for (int i = 0; i < 32; ++i) { buf[i] *= inv; gs += buf[i] * buf[i]; }
    red_s[tid] = gs;
    __syncthreads();
    for (int off = 128; off >= 1; off >>= 1) {
        if (tid < off) red_s[tid] += red_s[tid + off];
        __syncthreads();
    }
    float ginv = 1.0f / fmaxf(sqrtf(red_s[0]), 1e-12f);
    float* op = out + (size_t)n * (K_ * D_) + rid * 128 + d0;
#pragma unroll
    for (int i = 0; i < 32; ++i) op[i] = buf[i] * ginv;
}

extern "C" void kernel_launch(void* const* d_in, const int* in_sizes, int n_in,
                              void* d_out, int out_size, void* d_ws, size_t ws_size,
                              hipStream_t stream) {
    (void)in_sizes; (void)n_in; (void)out_size;
    const float* x    = (const float*)d_in[0];
    const float* cen  = (const float*)d_in[1];
    const float* wgt  = (const float*)d_in[2];
    const float* bia  = (const float*)d_in[3];
    const float* msk  = (const float*)d_in[4];
    float* out = (float*)d_out;

    // Largest C-split count whose partial buffers fit in workspace
    // (needed: N*S*(K*D + K) floats).  S=8 -> 512 WGs to saturate 23.3 TB/s.
    int S = 8;
    while (S > 1 && (size_t)N_ * S * (K_ * D_ + K_) * sizeof(float) > ws_size) S >>= 1;

    float* part  = (float*)d_ws;
    float* asump = part + (size_t)N_ * S * (K_ * D_);

    netvlad_partial<<<dim3(S, N_), dim3(256), 0, stream>>>(x, wgt, bia, msk, part, asump, S);
    netvlad_reduce<<<dim3(N_), dim3(256), 0, stream>>>(cen, part, asump, out, S);
}
